// GA_LSTMCell_59622736003905
// MI455X (gfx1250) — compile-verified, matrix operand order checked by reading, not run on hardware
//
#include <hip/hip_runtime.h>
#include <hip/hip_bf16.h>

typedef __attribute__((ext_vector_type(16))) _Float16 v16h;
typedef __attribute__((ext_vector_type(8)))  _Float16 v8h;
typedef __attribute__((ext_vector_type(8)))  float    v8f;

#define H_DIM 64
#define D_DIM 128
#define DH    (D_DIM + H_DIM)   // 192
#define G_DIM (4 * H_DIM)       // 256
#define KL    (2 * H_DIM)       // 128 = [h_agg | h_prev]

// Convert 8 consecutive f32 -> elements [off..off+7] of an f16 fragment.
// Emits 2x global_load_b128 + 4x v_cvt_pk_f16_f32.
static __device__ __forceinline__ void cvt8(const float* __restrict__ p, v16h& a, int off) {
    float4 f0 = *reinterpret_cast<const float4*>(p);
    float4 f1 = *reinterpret_cast<const float4*>(p + 4);
    a[off + 0] = (_Float16)f0.x; a[off + 1] = (_Float16)f0.y;
    a[off + 2] = (_Float16)f0.z; a[off + 3] = (_Float16)f0.w;
    a[off + 4] = (_Float16)f1.x; a[off + 5] = (_Float16)f1.y;
    a[off + 6] = (_Float16)f1.z; a[off + 7] = (_Float16)f1.w;
}

// Order-preserving float->uint encoding for atomic max.
static __device__ __forceinline__ unsigned encf(float f) {
    unsigned u = __float_as_uint(f);
    return (u & 0x80000000u) ? ~u : (u | 0x80000000u);
}
static __device__ __forceinline__ float decf(unsigned k) {
    unsigned u = (k & 0x80000000u) ? (k & 0x7FFFFFFFu) : ~k;
    return __uint_as_float(u);
}

// ---- weight staging: f16, transposed to [N][K] so B fragments are two b128 loads
__global__ void k_cvt_w(const float* __restrict__ W_gat, const float* __restrict__ W_ih,
                        const float* __restrict__ W_hh,
                        _Float16* __restrict__ WgT,    // [64][192]
                        _Float16* __restrict__ WlT) {  // [256][128]
    int t = blockIdx.x * blockDim.x + threadIdx.x;
    if (t < H_DIM * DH) {
        int n = t / DH, k = t - n * DH;
        WgT[t] = (_Float16)W_gat[(size_t)k * H_DIM + n];
    } else if (t < H_DIM * DH + G_DIM * KL) {
        int t2 = t - H_DIM * DH;
        int n = t2 >> 7, k = t2 & 127;
        float v = (k < H_DIM) ? W_ih[(size_t)k * G_DIM + n]
                              : W_hh[(size_t)(k - H_DIM) * G_DIM + n];
        WlT[t2] = (_Float16)v;
    }
}

// ---- init: h_agg = b_gat (bias pre-folded), emax/denom reset ----------------
__global__ void k_init(float* __restrict__ h_agg, const float* __restrict__ b_gat,
                       unsigned* __restrict__ emax_enc, float* __restrict__ denom, int Nn) {
    long t = (long)blockIdx.x * blockDim.x + threadIdx.x;
    if (t < (long)Nn * H_DIM) h_agg[t] = b_gat[t & (H_DIM - 1)];
    if (t < Nn) { emax_enc[t] = 0u; denom[t] = 0.f; }
}

// ---- kernel 1: z = [x | h_prev] @ W_gat via WMMA ----------------------------
__global__ void k_gat_gemm(const float* __restrict__ x, const float* __restrict__ h_prev,
                           const _Float16* __restrict__ WgT, float* __restrict__ z, int Nn) {
    int wave  = blockIdx.x * (blockDim.x >> 5) + (threadIdx.x >> 5);
    int lane  = threadIdx.x & 31;
    int mtile = wave >> 2;                 // 4 n-tiles (H=64)
    int ntile = wave & 3;
    int mtiles = (Nn + 15) >> 4;
    if (mtile >= mtiles) return;           // wave-uniform
    int r = lane & 15, hh = lane >> 4;
    int m  = mtile * 16 + r;
    int mc = m < Nn ? m : Nn - 1;
    const _Float16* bp = WgT + (size_t)(ntile * 16 + r) * DH;

    v8f c = {};
    #pragma unroll
    for (int kc = 0; kc < DH; kc += 32) {
        // concat boundary (128) is a chunk multiple -> source uniform per chunk
        const float* ap = (kc < D_DIM) ? (x + (size_t)mc * D_DIM + kc)
                                       : (h_prev + (size_t)mc * H_DIM + (kc - D_DIM));
        v16h a, b;
        cvt8(ap + 8 * hh,      a, 0);      // K = kc+8hh .. +7      (VGPR 0-3)
        cvt8(ap + 16 + 8 * hh, a, 8);      // K = kc+16+8hh .. +7   (VGPR 4-7)
        *reinterpret_cast<v8h*>(&b)       = *reinterpret_cast<const v8h*>(bp + kc + 8 * hh);
        *(reinterpret_cast<v8h*>(&b) + 1) = *reinterpret_cast<const v8h*>(bp + kc + 16 + 8 * hh);
        c = __builtin_amdgcn_wmma_f32_16x16x32_f16(false, a, false, b, (short)0, c,
                                                   false, false);
    }
    int n = ntile * 16 + r;
    if (mtile * 16 + 16 <= Nn) {           // full tile: unguarded stores
        #pragma unroll
        for (int q = 0; q < 8; ++q)
            z[(size_t)(mtile * 16 + q + 8 * hh) * H_DIM + n] = c[q];
    } else {
        #pragma unroll
        for (int q = 0; q < 8; ++q) {
            int mm = mtile * 16 + q + 8 * hh;
            if (mm < Nn) z[(size_t)mm * H_DIM + n] = c[q];
        }
    }
}

// ---- kernel 1b: a_s = z @ att_src, a_d = z @ att_dst ------------------------
__global__ void k_attn(const float* __restrict__ z, const float* __restrict__ att_src,
                       const float* __restrict__ att_dst, float* __restrict__ a_s,
                       float* __restrict__ a_d, int Nn) {
    int n = blockIdx.x * blockDim.x + threadIdx.x;
    if (n >= Nn) return;
    float s = 0.f, d = 0.f;
    #pragma unroll 8
    for (int j = 0; j < H_DIM; ++j) {
        float zv = z[(size_t)n * H_DIM + j];
        s += zv * att_src[j];
        d += zv * att_dst[j];
    }
    a_s[n] = s; a_d[n] = d;
}

// ---- kernel 2: e = LeakyReLU(a_s[src]+a_d[dst]); segment max ----------------
__global__ void k_edge_max(const int* __restrict__ ei, const float* __restrict__ a_s,
                           const float* __restrict__ a_d, float* __restrict__ ebuf,
                           unsigned* __restrict__ emax_enc, int Ee, int Nn) {
    long t = (long)blockIdx.x * blockDim.x + threadIdx.x;
    long total = (long)Ee + Nn;
    if (t >= total) return;
    int s, d;
    if (t < Ee) { s = ei[t]; d = ei[(long)Ee + t]; }
    else        { s = d = (int)(t - Ee); }           // self loop
    float e = a_s[s] + a_d[d];
    e = e > 0.f ? e : 0.2f * e;
    ebuf[t] = e;
    atomicMax(&emax_enc[d], encf(e));
}

// ---- kernel 3: ex = exp(e - emax[dst]); segment sum -------------------------
__global__ void k_edge_exp(const int* __restrict__ ei, const unsigned* __restrict__ emax_enc,
                           float* __restrict__ ebuf, float* __restrict__ denom,
                           int Ee, int Nn) {
    long t = (long)blockIdx.x * blockDim.x + threadIdx.x;
    long total = (long)Ee + Nn;
    if (t >= total) return;
    int d = (t < Ee) ? ei[(long)Ee + t] : (int)(t - Ee);
    float ex = __expf(ebuf[t] - decf(emax_enc[d]));
    ebuf[t] = ex;
    atomicAdd(&denom[d], ex);
}

// ---- kernel 4: h_agg[dst] += alpha * z[src] (1 wave / edge) -----------------
__global__ void k_edge_agg(const int* __restrict__ ei, const float* __restrict__ ebuf,
                           const float* __restrict__ denom, const float* __restrict__ z,
                           float* __restrict__ h_agg, int Ee, int Nn) {
    long wave = (long)blockIdx.x * (blockDim.x >> 5) + (threadIdx.x >> 5);
    int lane = threadIdx.x & 31;
    long total = (long)Ee + Nn;
    if (wave >= total) return;                       // wave-uniform
    int s, d;
    if (wave < Ee) { s = ei[wave]; d = ei[(long)Ee + wave]; }
    else           { s = d = (int)(wave - Ee); }
    const float* zr = z + (size_t)s * H_DIM + 2 * lane;
    __builtin_prefetch(zr, 0, 0);                    // global_prefetch_b8
    float alpha = ebuf[wave] / denom[d];
    float2 zv = *reinterpret_cast<const float2*>(zr);
    float* hr = h_agg + (size_t)d * H_DIM + 2 * lane;
    atomicAdd(hr,     alpha * zv.x);
    atomicAdd(hr + 1, alpha * zv.y);
}

// ---- kernel 5: gates GEMM (WMMA) + fused LSTM -------------------------------
// Block = 16 rows, 16 waves; each wave does one 16x16 tile of the 16x256 gate
// stripe; stripe lives in LDS, nonlinearity applied in-block (no HBM gates).
__global__ void k_lstm(const float* __restrict__ h_agg, const float* __restrict__ h_prev,
                       const float* __restrict__ c_prev, const _Float16* __restrict__ WlT,
                       const float* __restrict__ b_ih, const float* __restrict__ b_hh,
                       float* __restrict__ out, int Nn) {
    __shared__ float gates[16][G_DIM];               // 16 KB
    int mtile = blockIdx.x;
    int wv   = threadIdx.x >> 5;                     // 0..15 -> gate column tile
    int lane = threadIdx.x & 31;
    int r = lane & 15, hh = lane >> 4;
    int m  = mtile * 16 + r;
    int mc = m < Nn ? m : Nn - 1;
    const _Float16* bp = WlT + (size_t)(wv * 16 + r) * KL;

    v8f c = {};
    #pragma unroll
    for (int kc = 0; kc < KL; kc += 32) {            // K = 128: [h_agg | h_prev]
        const float* ap = (kc < H_DIM) ? (h_agg + (size_t)mc * H_DIM + kc)
                                       : (h_prev + (size_t)mc * H_DIM + (kc - H_DIM));
        v16h a, b;
        cvt8(ap + 8 * hh,      a, 0);
        cvt8(ap + 16 + 8 * hh, a, 8);
        *reinterpret_cast<v8h*>(&b)       = *reinterpret_cast<const v8h*>(bp + kc + 8 * hh);
        *(reinterpret_cast<v8h*>(&b) + 1) = *reinterpret_cast<const v8h*>(bp + kc + 16 + 8 * hh);
        c = __builtin_amdgcn_wmma_f32_16x16x32_f16(false, a, false, b, (short)0, c,
                                                   false, false);
    }
    int n = wv * 16 + r;
    float bias = b_ih[n] + b_hh[n];
    #pragma unroll
    for (int q = 0; q < 8; ++q) gates[q + 8 * hh][n] = c[q] + bias;
    __syncthreads();

    for (int e2 = threadIdx.x; e2 < 16 * H_DIM; e2 += blockDim.x) {
        int row = e2 >> 6, col = e2 & 63;
        int mm = mtile * 16 + row;
        if (mm >= Nn) continue;
        float ig = gates[row][col];
        float fg = gates[row][col + H_DIM];
        float gg = gates[row][col + 2 * H_DIM];
        float og = gates[row][col + 3 * H_DIM];
        float cp = c_prev[(size_t)mm * H_DIM + col];
        float si = 1.f / (1.f + __expf(-ig));
        float sf = 1.f / (1.f + __expf(-fg));
        float so = 1.f / (1.f + __expf(-og));
        float cn = sf * cp + si * tanhf(gg);
        float hn = so * tanhf(cn);
        out[(size_t)mm * H_DIM + col] = hn;                        // h_next
        out[(size_t)Nn * H_DIM + (size_t)mm * H_DIM + col] = cn;   // c_next
    }
}

extern "C" void kernel_launch(void* const* d_in, const int* in_sizes, int n_in,
                              void* d_out, int out_size, void* d_ws, size_t ws_size,
                              hipStream_t stream) {
    const float* x       = (const float*)d_in[0];
    const float* h_prev  = (const float*)d_in[1];
    const float* c_prev  = (const float*)d_in[2];
    const int*   ei      = (const int*)  d_in[3];
    const float* W_gat   = (const float*)d_in[4];
    const float* att_src = (const float*)d_in[5];
    const float* att_dst = (const float*)d_in[6];
    const float* b_gat   = (const float*)d_in[7];
    const float* W_ih    = (const float*)d_in[8];
    const float* b_ih    = (const float*)d_in[9];
    const float* W_hh    = (const float*)d_in[10];
    const float* b_hh    = (const float*)d_in[11];

    int Nn = in_sizes[1] / H_DIM;
    int Ee = in_sizes[3] / 2;
    long total = (long)Ee + Nn;

    // f16 transposed weights first (keeps 16B alignment for b128 loads)
    _Float16* WgT = (_Float16*)d_ws;                     // 64*192 halves (24576 B)
    _Float16* WlT = WgT + (size_t)H_DIM * DH;            // 256*128 halves (65536 B)
    float* ws = (float*)(WlT + (size_t)G_DIM * KL);
    float*    z        = ws;              ws += (size_t)Nn * H_DIM;
    float*    a_s      = ws;              ws += Nn;
    float*    a_d      = ws;              ws += Nn;
    float*    denom    = ws;              ws += Nn;
    unsigned* emax_enc = (unsigned*)ws;   ws += Nn;
    float*    ebuf     = ws;              ws += total;
    float*    h_agg    = ws;

    int mtiles = (Nn + 15) >> 4;

    int welems = H_DIM * DH + G_DIM * KL;                // 45056
    k_cvt_w<<<(unsigned)((welems + 255) / 256), 256, 0, stream>>>(W_gat, W_ih, W_hh,
                                                                  WgT, WlT);
    long initT = (long)Nn * H_DIM;
    k_init<<<(unsigned)((initT + 255) / 256), 256, 0, stream>>>(h_agg, b_gat, emax_enc,
                                                                denom, Nn);
    long waves1 = (long)mtiles * 4;
    k_gat_gemm<<<(unsigned)((waves1 + 7) / 8), 256, 0, stream>>>(x, h_prev, WgT, z, Nn);
    k_attn<<<(unsigned)((Nn + 255) / 256), 256, 0, stream>>>(z, att_src, att_dst,
                                                             a_s, a_d, Nn);
    k_edge_max<<<(unsigned)((total + 255) / 256), 256, 0, stream>>>(ei, a_s, a_d, ebuf,
                                                                    emax_enc, Ee, Nn);
    k_edge_exp<<<(unsigned)((total + 255) / 256), 256, 0, stream>>>(ei, emax_enc, ebuf,
                                                                    denom, Ee, Nn);
    k_edge_agg<<<(unsigned)((total + 7) / 8), 256, 0, stream>>>(ei, ebuf, denom, z,
                                                                h_agg, Ee, Nn);
    k_lstm<<<(unsigned)mtiles, 512, 0, stream>>>(h_agg, h_prev, c_prev, WlT, b_ih, b_hh,
                                                 c_prev == nullptr ? nullptr : (float*)d_out,
                                                 Nn);
}
